// DualStreamStegoDiSBlock_2061584302195
// MI455X (gfx1250) — matrix operand order repaired and checked by reading, not compiled, not run on hardware
//
#include <hip/hip_runtime.h>

// ---------------------------------------------------------------------------
// DualStreamStegoDiSBlock forward for MI455X (gfx1250, wave32, WMMA).
// ~150 GFLOP of dense projections vs ~300 MB traffic -> compute bound ->
// all GEMMs via v_wmma_f32_16x16x32_bf16 (bf16 operands, f32 accumulate).
// GEMM uses double-buffered LDS staging filled by global_load_async_to_lds_b128
// (ASYNCcnt) so global traffic overlaps WMMA. LayerNorm / conv / softplus /
// selective scan stay f32. Workspace requirement: ~310 MB.
// ---------------------------------------------------------------------------

#define BB      4
#define LL      1024
#define DIMC    1024
#define DIC     2048
#define TWO_DI  4096
#define BL      4096          // BB*LL
#define NX      96            // dt_rank + 2*d_state
#define DTR     64
#define NST     16
#define SEC     512

typedef __attribute__((ext_vector_type(16))) __bf16 bf16x16;
typedef __attribute__((ext_vector_type(8)))  float  f32x8;

// ------------------------------- GEMM --------------------------------------
// C[M,N] = A[M,K] (bf16 row-major) * B[N,K]^T (bf16 row-major) (+ bias[N])
// Block tile 128x64, 8 waves in a 4x2 grid; each wave owns a 32x32 tile
// (2x2 WMMA accumulators). K-chunks of 32, double-buffered async LDS staging.
#define LDSP 40   // padded LDS row (bf16): 80B rows, 16B-aligned chunks

__device__ __forceinline__ void async_ld16(unsigned lds_off, const void* g) {
  asm volatile("global_load_async_to_lds_b128 %0, %1, off"
               :: "v"(lds_off), "v"((unsigned long long)(uintptr_t)g)
               : "memory");
}
__device__ __forceinline__ void wait_async0() {
  asm volatile("s_wait_asynccnt 0x0" ::: "memory");
}

__global__ __launch_bounds__(256)
void gemm_bf16_wmma(const __bf16* __restrict__ A, const __bf16* __restrict__ Bw,
                    const float* __restrict__ bias, float* __restrict__ C,
                    int M, int N, int K) {
  __shared__ __bf16 As[2][128 * LDSP];
  __shared__ __bf16 Bs[2][64 * LDSP];
  const int t    = threadIdx.x;
  const int wave = t >> 5;
  const int lane = t & 31;
  const int m0   = blockIdx.y * 128;
  const int n0   = blockIdx.x * 64;
  const int wm   = (wave & 3) * 32;   // wave m offset in block tile
  const int wn   = (wave >> 2) * 32;  // wave n offset in block tile

  f32x8 acc00 = {}, acc01 = {}, acc10 = {}, acc11 = {};

  // staging maps (per thread): A needs 2x16B, B needs 1x16B per 32-K chunk
  auto stage = [&](int kc, int buf) {
#pragma unroll
    for (int j = 0; j < 2; ++j) {
      const int i = t + j * 256;
      const int r = i >> 2, c = (i & 3) * 8;          // 128 rows x 4 chunks
      const __bf16* g = A + (size_t)(m0 + r) * K + kc + c;
      async_ld16((unsigned)(uintptr_t)&As[buf][r * LDSP + c], g);
    }
    {
      const int r = t >> 2, c = (t & 3) * 8;          // 64 rows x 4 chunks
      int br = n0 + r; if (br > N - 1) br = N - 1;    // clamp OOB rows (cols >= N never stored)
      const __bf16* g = Bw + (size_t)br * K + kc + c;
      async_ld16((unsigned)(uintptr_t)&Bs[buf][r * LDSP + c], g);
    }
  };

  const int nk = K >> 5;
  stage(0, 0);
  wait_async0();
  __syncthreads();

  for (int ck = 0; ck < nk; ++ck) {
    const int buf = ck & 1;
    if (ck + 1 < nk) stage((ck + 1) << 5, buf ^ 1);   // overlap fill of other buffer

    union Frag { bf16x16 v; uint4 q[2]; } a0, a1, b0, b1;
    const __bf16* Ab = As[buf];
    const __bf16* Bb = Bs[buf];
    {
      // A 16x32 bf16 layout (ISA 7.12.2): lane&15 = M; lanes>=16 -> K offset +8
      const int mA   = wm + (lane & 15);
      const int koff = (lane & 16) >> 1;              // 0 or 8
      a0.q[0] = *(const uint4*)(Ab + mA * LDSP + koff);
      a0.q[1] = *(const uint4*)(Ab + mA * LDSP + 16 + koff);
      a1.q[0] = *(const uint4*)(Ab + (mA + 16) * LDSP + koff);
      a1.q[1] = *(const uint4*)(Ab + (mA + 16) * LDSP + 16 + koff);
    }
    {
      // B 32x16 bf16 layout: lane&15 = N; lanes>=16 -> K offset +8 (same as A)
      const int nB   = wn + (lane & 15);
      const int koff = (lane & 16) >> 1;              // 0 or 8
      b0.q[0] = *(const uint4*)(Bb + nB * LDSP + koff);
      b0.q[1] = *(const uint4*)(Bb + nB * LDSP + 16 + koff);
      b1.q[0] = *(const uint4*)(Bb + (nB + 16) * LDSP + koff);
      b1.q[1] = *(const uint4*)(Bb + (nB + 16) * LDSP + 16 + koff);
    }
    acc00 = __builtin_amdgcn_wmma_f32_16x16x32_bf16(false, a0.v, false, b0.v, (short)0, acc00, false, false);
    acc01 = __builtin_amdgcn_wmma_f32_16x16x32_bf16(false, a0.v, false, b1.v, (short)0, acc01, false, false);
    acc10 = __builtin_amdgcn_wmma_f32_16x16x32_bf16(false, a1.v, false, b0.v, (short)0, acc10, false, false);
    acc11 = __builtin_amdgcn_wmma_f32_16x16x32_bf16(false, a1.v, false, b1.v, (short)0, acc11, false, false);

    wait_async0();        // next buffer fully in LDS
    __syncthreads();      // everyone done reading current buffer
  }

  // C/D layout: vgpr r -> M = r (+8 for lanes 16..31); lane&15 = N
  const int mb = m0 + wm + ((lane & 16) >> 1);
  const int nb = n0 + wn + (lane & 15);
  auto store_tile = [&](const f32x8& acc, int mt, int nt) {
    if (nt < N) {
      const float bv = bias ? bias[nt] : 0.f;
#pragma unroll
      for (int r = 0; r < 8; ++r)
        C[(size_t)(mt + r) * N + nt] = acc[r] + bv;
    }
  };
  store_tile(acc00, mb,      nb);
  store_tile(acc01, mb,      nb + 16);
  store_tile(acc10, mb + 16, nb);
  store_tile(acc11, mb + 16, nb + 16);
}

// --------------------------- elementwise kernels ---------------------------
__global__ void cvt_bf16_kernel(const float* __restrict__ in,
                                __bf16* __restrict__ out, int n) {
  int i = blockIdx.x * 256 + threadIdx.x;
  if (i < n) out[i] = (__bf16)in[i];
}

__global__ void cvt_cols_kernel(const float* __restrict__ in,
                                __bf16* __restrict__ out,
                                int rows, int cols_in, int cols_out) {
  int i = blockIdx.x * 256 + threadIdx.x;
  if (i >= rows * cols_out) return;
  int r = i / cols_out, c = i - r * cols_out;
  out[i] = (__bf16)in[(size_t)r * cols_in + c];
}

__global__ void silu_kernel(const float* __restrict__ in, float* __restrict__ out, int n) {
  int i = blockIdx.x * 256 + threadIdx.x;
  if (i < n) { float x = in[i]; out[i] = x / (1.f + __expf(-x)); }
}

__global__ void sigmoid_kernel(float* __restrict__ x, int n) {
  int i = blockIdx.x * 256 + threadIdx.x;
  if (i < n) x[i] = 1.f / (1.f + __expf(-x[i]));
}

// tiny dense (adaLN): out[Bn,N] = x[Bn,K] @ W[N,K]^T + bias
__global__ void dense_small_kernel(const float* __restrict__ x, const float* __restrict__ W,
                                   const float* __restrict__ bias, float* __restrict__ out,
                                   int Bn, int N, int K) {
  int i = blockIdx.x * 256 + threadIdx.x;
  if (i >= Bn * N) return;
  int b = i / N, n = i - b * N;
  const float* xr = x + (size_t)b * K;
  const float* wr = W + (size_t)n * K;
  float acc = bias ? bias[n] : 0.f;
  for (int k = 0; k < K; ++k) acc += xr[k] * wr[k];
  out[i] = acc;
}

// LayerNorm + adaLN modulate: one block per (b,l) row of DIM=1024
__global__ __launch_bounds__(256)
void modulate_kernel(const float* __restrict__ h, const float* __restrict__ nw,
                     const float* __restrict__ nb, const float* __restrict__ mod6,
                     float* __restrict__ out_f32, __bf16* __restrict__ out_bf16) {
  const int row = blockIdx.x;          // 0..BL-1
  const int b   = row >> 10;
  const float* x = h + (size_t)row * DIMC;
  __shared__ float red[256];
  float v[4]; float s = 0.f;
  for (int i = 0; i < 4; ++i) { v[i] = x[threadIdx.x + 256 * i]; s += v[i]; }
  red[threadIdx.x] = s; __syncthreads();
  for (int off = 128; off > 0; off >>= 1) {
    if (threadIdx.x < off) red[threadIdx.x] += red[threadIdx.x + off];
    __syncthreads();
  }
  const float mean = red[0] * (1.f / DIMC);
  __syncthreads();
  float s2 = 0.f;
  for (int i = 0; i < 4; ++i) { float d = v[i] - mean; s2 += d * d; }
  red[threadIdx.x] = s2; __syncthreads();
  for (int off = 128; off > 0; off >>= 1) {
    if (threadIdx.x < off) red[threadIdx.x] += red[threadIdx.x + off];
    __syncthreads();
  }
  const float rstd = rsqrtf(red[0] * (1.f / DIMC) + 1e-5f);
  const float* shift = mod6 + (size_t)b * 3 * DIMC;
  const float* scale = shift + DIMC;
  for (int i = 0; i < 4; ++i) {
    int j = threadIdx.x + 256 * i;
    float ln = (v[i] - mean) * rstd * nw[j] + nb[j];
    float o  = ln * (1.f + scale[j]) + shift[j];
    if (out_f32)  out_f32[(size_t)row * DIMC + j]  = o;
    if (out_bf16) out_bf16[(size_t)row * DIMC + j] = (__bf16)o;
  }
}

// depthwise causal conv (K=4) + silu, with optional sequence flip at the input
__global__ void conv_silu_kernel(const float* __restrict__ xz,   // BL x 4096, xd = cols 0..2047
                                 const float* __restrict__ cw,   // DI x 4
                                 const float* __restrict__ cb,   // DI
                                 int rev,
                                 float* __restrict__ u_f32,
                                 __bf16* __restrict__ u_bf16) {
  int idx = blockIdx.x * 256 + threadIdx.x;
  if (idx >= BL * DIC) return;
  const int d = idx & (DIC - 1);
  const int l = (idx >> 11) & (LL - 1);
  const int b = idx >> 21;
  float acc = cb[d];
  for (int j = 0; j < 4; ++j) {
    int li = l - 3 + j;
    if (li >= 0) {
      int ls = rev ? (LL - 1 - li) : li;
      acc += cw[d * 4 + j] * xz[((size_t)(b * LL + ls)) * TWO_DI + d];
    }
  }
  float u = acc / (1.f + __expf(-acc));
  u_f32[idx]  = u;
  u_bf16[idx] = (__bf16)u;
}

__global__ void bias_softplus_kernel(float* __restrict__ x, const float* __restrict__ bias, int n) {
  int i = blockIdx.x * 256 + threadIdx.x;
  if (i >= n) return;
  float v = x[i] + bias[i & (DIC - 1)];
  x[i] = fmaxf(v, 0.f) + log1pf(__expf(-fabsf(v)));
}

// selective scan: one thread per (b,d) channel, 16 SSM states in registers,
// B/C (32 floats/step) double-buffered through LDS: 1 barrier per timestep
__global__ __launch_bounds__(256)
void scan_kernel(const float* __restrict__ u, const float* __restrict__ delta,
                 const float* __restrict__ dbl,   // BL x 96 ; B at +64, C at +80
                 const float* __restrict__ A_log, // DI x 16
                 const float* __restrict__ Dp,    // DI
                 float* __restrict__ y) {
  const int d = blockIdx.x * 256 + threadIdx.x;
  const int b = blockIdx.y;
  float a[NST], h[NST];
  for (int n = 0; n < NST; ++n) { a[n] = -__expf(A_log[d * NST + n]); h[n] = 0.f; }
  const float Dd = Dp[d];
  __shared__ float sBC[2][32];
  if (threadIdx.x < 32)
    sBC[0][threadIdx.x] = dbl[(size_t)b * LL * NX + DTR + threadIdx.x];
  __syncthreads();
  for (int l = 0; l < LL; ++l) {
    const size_t row = (size_t)b * LL + l;
    const float* bc = sBC[l & 1];
    if (l + 1 < LL && threadIdx.x < 32)
      sBC[(l + 1) & 1][threadIdx.x] = dbl[(row + 1) * NX + DTR + threadIdx.x];
    const float dt = delta[row * DIC + d];
    const float ut = u[row * DIC + d];
    const float du = dt * ut;
    float yv = 0.f;
    for (int n = 0; n < NST; ++n) {
      h[n] = __expf(dt * a[n]) * h[n] + du * bc[n];
      yv  += h[n] * bc[NST + n];
    }
    y[row * DIC + d] = yv + ut * Dd;
    __syncthreads();
  }
}

__global__ void flip_acc_kernel(float* __restrict__ yacc, const float* __restrict__ ydir,
                                int rev, int first) {
  int idx = blockIdx.x * 256 + threadIdx.x;
  if (idx >= BL * DIC) return;
  const int d = idx & (DIC - 1);
  const int l = (idx >> 11) & (LL - 1);
  const int b = idx >> 21;
  const int ls = rev ? (LL - 1 - l) : l;
  float v = ydir[((size_t)(b * LL + ls)) * DIC + d];
  if (first) yacc[idx] = v; else yacc[idx] += v;
}

__global__ void ygz_kernel(const float* __restrict__ yacc, const float* __restrict__ xz,
                           __bf16* __restrict__ out) {
  int idx = blockIdx.x * 256 + threadIdx.x;
  if (idx >= BL * DIC) return;
  const int d = idx & (DIC - 1);
  const size_t row = (size_t)(idx >> 11);
  float z = xz[row * TWO_DI + DIC + d];
  float sil = z / (1.f + __expf(-z));
  out[idx] = (__bf16)(yacc[idx] * sil);
}

__global__ void x_make_kernel(const float* __restrict__ tex_mod, const float* __restrict__ sx,
                              const float* __restrict__ ax, __bf16* __restrict__ out) {
  int idx = blockIdx.x * 256 + threadIdx.x;
  if (idx >= BL * DIMC) return;
  out[idx] = (__bf16)(tex_mod[idx] + ax[0] * tanhf(sx[idx]));
}

__global__ void z_make_kernel(float* __restrict__ z, const float* __restrict__ g2,
                              const float* __restrict__ az) {
  int idx = blockIdx.x * 256 + threadIdx.x;
  if (idx >= BL * DIMC) return;
  z[idx] += az[0] * tanhf(g2[idx]);
}

__global__ void sem_final_kernel(const float* __restrict__ h, const float* __restrict__ mod6,
                                 const float* __restrict__ mo, float* __restrict__ out) {
  int idx = blockIdx.x * 256 + threadIdx.x;
  if (idx >= BL * DIMC) return;
  const int j = idx & (DIMC - 1);
  const int b = idx >> 20;
  out[idx] = h[idx] + mod6[(size_t)b * 3 * DIMC + 2 * DIMC + j] * mo[idx];
}

__global__ void tex_final_kernel(const float* __restrict__ h, const float* __restrict__ mod6,
                                 const float* __restrict__ mo, const float* __restrict__ z,
                                 const float* __restrict__ mask, float* __restrict__ out) {
  int idx = blockIdx.x * 256 + threadIdx.x;
  if (idx >= BL * DIMC) return;
  const int j = idx & (DIMC - 1);
  const int b = idx >> 20;
  float zz = z[idx];
  float sil = zz / (1.f + __expf(-zz));
  out[idx] = h[idx] + mod6[(size_t)b * 3 * DIMC + 2 * DIMC + j] * (mo[idx] * sil * mask[idx]);
}

// ------------------------------ host side ----------------------------------
static inline dim3 elg(long n) { return dim3((unsigned)((n + 255) / 256)); }

struct Ws {
  float *c_act, *mod6_s, *mod6_t, *tex_mod, *xz, *u_f32, *dbl, *delta,
        *y_dir, *y_acc, *m_out, *mask, *z_tex, *sx;
  __bf16 *act_a, *act_b, *wtb, *dtb;
};

static void gemm(hipStream_t s, const __bf16* A, const __bf16* Bw, const float* bias,
                 float* C, int M, int N, int K) {
  dim3 grid((N + 63) / 64, M / 128);
  gemm_bf16_wmma<<<grid, 256, 0, s>>>(A, Bw, bias, C, M, N, K);
}

static void run_mamba(hipStream_t s, void* const* d_in, int bs,
                      const __bf16* x_bf16, float* out, const Ws& w) {
  auto F = [&](int i) { return (const float*)d_in[i]; };
  // xz = x @ in_proj_w.T  : (4096 x 1024) x (4096 x 1024)^T
  cvt_bf16_kernel<<<elg(TWO_DI * DIMC), 256, 0, s>>>(F(bs + 0), w.wtb, TWO_DI * DIMC);
  gemm(s, x_bf16, w.wtb, nullptr, w.xz, BL, TWO_DI, DIMC);
  for (int dir = 0; dir < 2; ++dir) {
    const int o = bs + 2 + dir * 7;  // conv_w, conv_b, xproj_w, dt_w, dt_bias, A_log, D
    conv_silu_kernel<<<elg((long)BL * DIC), 256, 0, s>>>(w.xz, F(o + 0), F(o + 1), dir,
                                                         w.u_f32, w.act_a);
    cvt_bf16_kernel<<<elg(NX * DIC), 256, 0, s>>>(F(o + 2), w.wtb, NX * DIC);
    gemm(s, w.act_a, w.wtb, nullptr, w.dbl, BL, NX, DIC);            // dbl = u @ xproj^T
    cvt_cols_kernel<<<elg(BL * DTR), 256, 0, s>>>(w.dbl, w.dtb, BL, NX, DTR);
    cvt_bf16_kernel<<<elg(DIC * DTR), 256, 0, s>>>(F(o + 3), w.wtb, DIC * DTR);
    gemm(s, w.dtb, w.wtb, nullptr, w.delta, BL, DIC, DTR);           // dt @ dt_w^T
    bias_softplus_kernel<<<elg((long)BL * DIC), 256, 0, s>>>(w.delta, F(o + 4), BL * DIC);
    scan_kernel<<<dim3(DIC / 256, BB), 256, 0, s>>>(w.u_f32, w.delta, w.dbl,
                                                    F(o + 5), F(o + 6), w.y_dir);
    flip_acc_kernel<<<elg((long)BL * DIC), 256, 0, s>>>(w.y_acc, w.y_dir, dir, dir == 0);
  }
  ygz_kernel<<<elg((long)BL * DIC), 256, 0, s>>>(w.y_acc, w.xz, w.act_a);
  cvt_bf16_kernel<<<elg(DIMC * DIC), 256, 0, s>>>(F(bs + 1), w.wtb, DIMC * DIC);
  gemm(s, w.act_a, w.wtb, nullptr, out, BL, DIMC, DIC);              // out_proj
}

extern "C" void kernel_launch(void* const* d_in, const int* in_sizes, int n_in,
                              void* d_out, int out_size, void* d_ws, size_t ws_size,
                              hipStream_t stream) {
  (void)in_sizes; (void)n_in; (void)out_size; (void)ws_size;
  auto F = [&](int i) { return (const float*)d_in[i]; };
  // inputs (flattened setup dict order): 0 h_sem, 1 h_tex, 2 c_global,
  // 3 secret_seq, 4..21 params, 22..37 sem_mamba, 38..53 tex_mamba
  const float *h_sem = F(0), *h_tex = F(1), *c_glob = F(2), *secret = F(3);

  char* p = (char*)d_ws;
  auto alloc = [&](size_t bytes) { char* r = p; p += (bytes + 255) & ~(size_t)255; return r; };
  Ws w;
  w.c_act   = (float*)alloc((size_t)BB * DIMC * 4);
  w.mod6_s  = (float*)alloc((size_t)BB * 3 * DIMC * 4);
  w.mod6_t  = (float*)alloc((size_t)BB * 3 * DIMC * 4);
  w.tex_mod = (float*)alloc((size_t)BL * DIMC * 4);
  w.xz      = (float*)alloc((size_t)BL * TWO_DI * 4);
  w.u_f32   = (float*)alloc((size_t)BL * DIC * 4);
  w.dbl     = (float*)alloc((size_t)BL * NX * 4);
  w.delta   = (float*)alloc((size_t)BL * DIC * 4);
  w.y_dir   = (float*)alloc((size_t)BL * DIC * 4);
  w.y_acc   = (float*)alloc((size_t)BL * DIC * 4);
  w.m_out   = (float*)alloc((size_t)BL * DIMC * 4);
  w.mask    = (float*)alloc((size_t)BL * DIMC * 4);
  w.z_tex   = (float*)alloc((size_t)BL * DIMC * 4);
  w.sx      = (float*)alloc((size_t)BL * DIMC * 4);
  w.act_a   = (__bf16*)alloc((size_t)BL * DIC * 2);
  w.act_b   = (__bf16*)alloc((size_t)BL * DIMC * 2);
  w.wtb     = (__bf16*)alloc((size_t)TWO_DI * DIMC * 2);
  w.dtb     = (__bf16*)alloc((size_t)BL * DTR * 2);

  float* out_sem = (float*)d_out;
  float* out_tex = out_sem + (size_t)BL * DIMC;

  // conditioning: c = silu(c_global); adaLN shift/scale/gate for both streams
  silu_kernel<<<elg(BB * DIMC), 256, 0, stream>>>(c_glob, w.c_act, BB * DIMC);
  dense_small_kernel<<<elg(BB * 3 * DIMC), 256, 0, stream>>>(w.c_act, F(8), F(9), w.mod6_s,
                                                             BB, 3 * DIMC, DIMC);
  dense_small_kernel<<<elg(BB * 3 * DIMC), 256, 0, stream>>>(w.c_act, F(10), F(11), w.mod6_t,
                                                             BB, 3 * DIMC, DIMC);

  // ---- semantic stream ----
  modulate_kernel<<<BL, 256, 0, stream>>>(h_sem, F(4), F(5), w.mod6_s, nullptr, w.act_b);
  run_mamba(stream, d_in, 22, w.act_b, w.m_out, w);
  sem_final_kernel<<<elg((long)BL * DIMC), 256, 0, stream>>>(h_sem, w.mod6_s, w.m_out, out_sem);

  // mask = sigmoid(h_sem_out @ mask_proj.T + b)
  cvt_bf16_kernel<<<elg((long)BL * DIMC), 256, 0, stream>>>(out_sem, w.act_b, BL * DIMC);
  cvt_bf16_kernel<<<elg(DIMC * DIMC), 256, 0, stream>>>(F(12), w.wtb, DIMC * DIMC);
  gemm(stream, w.act_b, w.wtb, F(13), w.mask, BL, DIMC, DIMC);
  sigmoid_kernel<<<elg((long)BL * DIMC), 256, 0, stream>>>(w.mask, BL * DIMC);

  // ---- texture stream ----
  modulate_kernel<<<BL, 256, 0, stream>>>(h_tex, F(6), F(7), w.mod6_t, w.tex_mod, w.act_b);
  cvt_bf16_kernel<<<elg((long)BL * SEC), 256, 0, stream>>>(secret, w.act_a, BL * SEC);
  cvt_bf16_kernel<<<elg(DIMC * SEC), 256, 0, stream>>>(F(16), w.wtb, DIMC * SEC);
  gemm(stream, w.act_a, w.wtb, F(17), w.sx, BL, DIMC, SEC);          // secret -> x
  cvt_bf16_kernel<<<elg(DIMC * DIMC), 256, 0, stream>>>(F(14), w.wtb, DIMC * DIMC);
  gemm(stream, w.act_b, w.wtb, F(15), w.z_tex, BL, DIMC, DIMC);      // tex_to_z
  cvt_bf16_kernel<<<elg(DIMC * SEC), 256, 0, stream>>>(F(18), w.wtb, DIMC * SEC);
  gemm(stream, w.act_a, w.wtb, F(19), w.delta, BL, DIMC, SEC);       // secret -> z (g2)
  x_make_kernel<<<elg((long)BL * DIMC), 256, 0, stream>>>(w.tex_mod, w.sx, F(20), w.act_b);
  z_make_kernel<<<elg((long)BL * DIMC), 256, 0, stream>>>(w.z_tex, w.delta, F(21));
  run_mamba(stream, d_in, 38, w.act_b, w.m_out, w);
  tex_final_kernel<<<elg((long)BL * DIMC), 256, 0, stream>>>(h_tex, w.mod6_t, w.m_out,
                                                             w.z_tex, w.mask, out_tex);
}
